// InceptionAuxQ_51479478010584
// MI455X (gfx1250) — matrix operand order repaired and checked by reading, not compiled, not run on hardware
//
#include <hip/hip_runtime.h>
#include <hip/hip_bf16.h>
#include <stddef.h>

// ---------------------------------------------------------------------------
// CDNA5 (gfx1250) fused quantized Inception-aux head.
//   stage P: avgpool 5x5 s3 (256,768,17,17)->(256,768,5,5), emit f16 A1[6400][768]
//   stage 1: WMMA GEMM [6400,768]x[768,128] + BN0/ReLU/PACT-quant -> f16 X1
//   stage 2: WMMA GEMM [256,3200]x[3200,768] + BN1/ReLU/PACT-quant -> f16 X2
//   stage 3: WMMA GEMM [256,768]x[768,1024(pad)] + bias -> float d_out[256,1000]
//
// Data path: no LDS in the GEMMs. A fragments are two aligned uint4 loads from
// row-major f16; B is pre-packed as Bf[k/32][n][32] so each lane's fragment is
// two aligned uint4 loads. Each wave computes a 16x64 tile (A frag reused 4x).
// ---------------------------------------------------------------------------

typedef __attribute__((ext_vector_type(16))) _Float16 v16h;
typedef __attribute__((ext_vector_type(8)))  float    v8f;

union Frag16 { v16h v; unsigned int u[8]; uint4 q[2]; };

#define BN_EPS 1e-3f

// ---------------- weight scale: single-block abs-max reduction -------------
__global__ __launch_bounds__(256) void absmax_scale_kernel(const float* __restrict__ w,
                                                           int n, float qmax,
                                                           float* __restrict__ scale) {
    __shared__ float red[256];
    float m = 0.f;
    for (int i = threadIdx.x; i < n; i += 256) m = fmaxf(m, fabsf(w[i]));
    red[threadIdx.x] = m;
    __syncthreads();
    for (int s = 128; s > 0; s >>= 1) {
        if ((int)threadIdx.x < s) red[threadIdx.x] = fmaxf(red[threadIdx.x], red[threadIdx.x + s]);
        __syncthreads();
    }
    if (threadIdx.x == 0) scale[0] = red[0] / qmax;
}

__device__ __forceinline__ float qdq_w(float x, float s) {
    float q = rintf(x / s);
    q = fminf(fmaxf(q, -127.f), 127.f);
    return q * s;
}

// quantize + repack: w[N][K] float -> Bf[(k>>5)*Npad + n][k&31] f16 (fragment-native)
__global__ __launch_bounds__(256) void quant_pack_kernel(const float* __restrict__ w,
                                                         const float* __restrict__ scale,
                                                         _Float16* __restrict__ outB,
                                                         int N, int K, int Npad) {
    int idx = blockIdx.x * 256 + threadIdx.x;
    if (idx >= K * Npad) return;
    int k = idx / Npad, n = idx - k * Npad;
    float v = 0.f;
    if (n < N) v = qdq_w(w[(size_t)n * K + k], scale[0]);
    outB[((size_t)(k >> 5) * Npad + n) * 32 + (k & 31)] = (_Float16)v;
}

// conv1 weights [co=768][ci=128][5][5] -> Bf with k = p*128+ci, n = co
__global__ __launch_bounds__(256) void quant_conv1_kernel(const float* __restrict__ w,
                                                          const float* __restrict__ scale,
                                                          _Float16* __restrict__ outB) {
    int idx = blockIdx.x * 256 + threadIdx.x;
    if (idx >= 3200 * 768) return;
    int k = idx / 768, co = idx - k * 768;
    int p = k >> 7, ci = k & 127;                     // k = p*128 + ci
    float v = qdq_w(w[(size_t)co * 3200 + ci * 25 + p], scale[0]);
    outB[((size_t)(k >> 5) * 768 + co) * 32 + (k & 31)] = (_Float16)v;
}

// avgpool 5x5 s3 VALID, NCHW -> A1[(b*25 + h*5 + w)][c] f16 (row-major, K=768)
__global__ __launch_bounds__(256) void avgpool_kernel(const float* __restrict__ act,
                                                      _Float16* __restrict__ out) {
    int idx = blockIdx.x * 256 + threadIdx.x;
    if (idx >= 6400 * 768) return;
    int c = idx % 768;
    int r = idx / 768;            // b*25 + p
    int p = r % 25, b = r / 25;
    int h0 = (p / 5) * 3, w0 = (p % 5) * 3;
    const float* base = act + (((size_t)b * 768 + c) * 17 + h0) * 17 + w0;
    float s = 0.f;
#pragma unroll
    for (int i = 0; i < 5; ++i)
#pragma unroll
        for (int j = 0; j < 5; ++j) s += base[i * 17 + j];
    out[idx] = (_Float16)(s * 0.04f);
}

// ---------------- WMMA GEMM, LDS-free, fused epilogue ----------------------
// Block = 128 threads = 4 waves. Each wave computes a 16(M) x 64(N) tile:
// one A fragment per K-step feeds 4 WMMAs (4 accumulators / 4 N slabs).
// MODE 0: BN -> ReLU -> PACT quant(alpha, 255 levels) -> f16 outh[ldo]
// MODE 1: + bias[n] -> float outf[nlim] (cols < nlim only)
template <int MODE>
__global__ __launch_bounds__(128) void gemm_wmma_fused(const _Float16* __restrict__ A,
                                                       const _Float16* __restrict__ Bf,
                                                       int K, int Npad,
                                                       const float* __restrict__ gamma,
                                                       const float* __restrict__ beta,
                                                       const float* __restrict__ mean,
                                                       const float* __restrict__ var,
                                                       const float* __restrict__ alphap,
                                                       const float* __restrict__ bias,
                                                       _Float16* __restrict__ outh, int ldo,
                                                       float* __restrict__ outf, int nlim) {
    const int tid  = threadIdx.x;
    const int lane = tid & 31;
    const int wv   = tid >> 5;
    const int half = lane >> 4;
    const int mr   = lane & 15;

    const int m0 = (blockIdx.x * 4 + wv) * 16;   // this wave's 16 rows
    const int n0 = blockIdx.y * 64;              // this block's 64 cols

    v8f acc[4] = {{}, {}, {}, {}};

    // per-lane row base for A fragment loads (row m0+mr, f16 row-major, stride K)
    const _Float16* aRow = A + (size_t)(m0 + mr) * K;

    for (int k0 = 0; k0 < K; k0 += 32) {
        // ---- A fragment: two aligned 16B chunks of row (m0+mr) ------------
        // af.u[0..3] <- halves [k0 + 8*half .. +7]; af.u[4..7] <- +16 further
        Frag16 af;
        const uint4* ap = reinterpret_cast<const uint4*>(aRow + k0 + 8 * half);
        af.q[0] = ap[0];
        af.q[1] = ap[2];                         // +16 halves = +2 uint4
        if (k0 + 32 < K)                         // hint next A tile (global_prefetch)
            __builtin_prefetch(aRow + k0 + 32, 0, 1);

        // ---- 4 B fragments + 4 WMMAs --------------------------------------
        const _Float16* bTile = Bf + ((size_t)(k0 >> 5) * Npad) * 32;
#pragma unroll
        for (int t = 0; t < 4; ++t) {
            const int n = n0 + t * 16 + mr;
            // bf.q[0] <- B[k0 + 8*half .. +7][n]; bf.q[1] <- +16 further (matches A)
            Frag16 bf;
            const uint4* bp = reinterpret_cast<const uint4*>(bTile + (size_t)n * 32 + half * 8);
            bf.q[0] = bp[0];
            bf.q[1] = bp[2];
            acc[t] = __builtin_amdgcn_wmma_f32_16x16x32_f16(false, af.v, false, bf.v,
                                                            (short)0, acc[t], false, false);
        }
    }

    // ---- fused epilogue ---------------------------------------------------
#pragma unroll
    for (int t = 0; t < 4; ++t) {
        const int n = n0 + t * 16 + mr;
        if (MODE == 0) {
            float inv = gamma[n] * rsqrtf(var[n] + BN_EPS);
            float sh  = beta[n] - mean[n] * inv;
            float a   = alphap[0];
            float s   = a * (1.f / 255.f);
#pragma unroll
            for (int v = 0; v < 8; ++v) {
                int m = m0 + v + 8 * half;
                float x = acc[t][v] * inv + sh;      // BN
                x = fminf(fmaxf(x, 0.f), a);         // ReLU + clip to [0, alpha]
                x = rintf(x / s) * s;                // PACT uniform quant (255 levels)
                outh[(size_t)m * ldo + n] = (_Float16)x;
            }
        } else {
            if (n < nlim) {
                float bb = bias[n];
#pragma unroll
                for (int v = 0; v < 8; ++v) {
                    int m = m0 + v + 8 * half;
                    outf[(size_t)m * nlim + n] = acc[t][v] + bb;
                }
            }
        }
    }
}

// ---------------------------------------------------------------------------
extern "C" void kernel_launch(void* const* d_in, const int* in_sizes, int n_in,
                              void* d_out, int out_size, void* d_ws, size_t ws_size,
                              hipStream_t stream) {
    const float* act      = (const float*)d_in[0];
    // d_in[1] = alpha_in (unused by reference)
    const float* conv0_w  = (const float*)d_in[2];
    const float* bn0_g    = (const float*)d_in[3];
    const float* bn0_b    = (const float*)d_in[4];
    const float* bn0_m    = (const float*)d_in[5];
    const float* bn0_v    = (const float*)d_in[6];
    const float* alpha0   = (const float*)d_in[7];
    const float* conv1_w  = (const float*)d_in[8];
    const float* bn1_g    = (const float*)d_in[9];
    const float* bn1_b    = (const float*)d_in[10];
    const float* bn1_m    = (const float*)d_in[11];
    const float* bn1_v    = (const float*)d_in[12];
    const float* alpha1   = (const float*)d_in[13];
    const float* fc_w     = (const float*)d_in[14];
    const float* fc_b     = (const float*)d_in[15];
    float*       out      = (float*)d_out;

    char* ws = (char*)d_ws;
    // workspace layout (all 256B aligned)
    float*     scales = (float*)(ws + 0);                       // 3 floats
    _Float16*  B1     = (_Float16*)(ws + 256);                  // 768/32 x 128 x 32
    _Float16*  B2     = (_Float16*)(ws + 196864);               // 3200/32 x 768 x 32
    _Float16*  B3     = (_Float16*)(ws + 5112064);              // 768/32 x 1024 x 32
    _Float16*  A1     = (_Float16*)(ws + 6684928);              // 6400 x 768
    _Float16*  X1     = (_Float16*)(ws + 16515328);             // 6400 x 128
    _Float16*  X2     = (_Float16*)(ws + 18153728);             // 256 x 768

    // 1) per-tensor weight scales (max|w|/127)
    absmax_scale_kernel<<<1, 256, 0, stream>>>(conv0_w, 128 * 768,       127.f, scales + 0);
    absmax_scale_kernel<<<1, 256, 0, stream>>>(conv1_w, 768 * 128 * 25,  127.f, scales + 1);
    absmax_scale_kernel<<<1, 256, 0, stream>>>(fc_w,    1000 * 768,      127.f, scales + 2);

    // 2) quant-dequant weights into fragment-native f16 Bf matrices
    quant_pack_kernel<<<(768 * 128 + 255) / 256, 256, 0, stream>>>(
        conv0_w, scales + 0, B1, /*N=*/128, /*K=*/768, /*Npad=*/128);
    quant_conv1_kernel<<<(3200 * 768 + 255) / 256, 256, 0, stream>>>(conv1_w, scales + 1, B2);
    quant_pack_kernel<<<(768 * 1024 + 255) / 256, 256, 0, stream>>>(
        fc_w, scales + 2, B3, /*N=*/1000, /*K=*/768, /*Npad=*/1024);

    // 3) avgpool -> f16 A1 [6400][768]
    avgpool_kernel<<<(6400 * 768 + 255) / 256, 256, 0, stream>>>(act, A1);

    // 4) stage 1 GEMM: [6400,768] x [768,128] + BN0/ReLU/quant -> X1
    {
        dim3 grid(6400 / 64, 128 / 64);
        gemm_wmma_fused<0><<<grid, 128, 0, stream>>>(A1, B1, 768, 128,
            bn0_g, bn0_b, bn0_m, bn0_v, alpha0, nullptr, X1, 128, nullptr, 0);
    }
    // 5) stage 2 GEMM: X1 viewed as [256,3200] x [3200,768] + BN1/ReLU/quant -> X2
    {
        dim3 grid(256 / 64, 768 / 64);
        gemm_wmma_fused<0><<<grid, 128, 0, stream>>>(X1, B2, 3200, 768,
            bn1_g, bn1_b, bn1_m, bn1_v, alpha1, nullptr, X2, 768, nullptr, 0);
    }
    // 6) stage 3 GEMM: [256,768] x [768,1024] + bias -> float out [256,1000]
    {
        dim3 grid(256 / 64, 1024 / 64);
        gemm_wmma_fused<1><<<grid, 128, 0, stream>>>(X2, B3, 768, 1024,
            nullptr, nullptr, nullptr, nullptr, nullptr, fc_b, nullptr, 0, out, 1000);
    }
}
